// SwinTransformerBlock_42167988912267
// MI455X (gfx1250) — matrix operand order repaired and checked by reading, not compiled, not run on hardware
//
#include <hip/hip_runtime.h>
#include <cstdint>
#include <cstddef>

// ---------------------------------------------------------------- constants
#define BB      32
#define HH      64
#define CC      128
#define HEADS   4
#define HD      32
#define WS_     8
#define SS_     4
#define NTOK    64          // tokens per window
#define NWIN    2048        // B * (H/WS)*(W/WS)
#define MLP_HID 512
#define SCALE_  0.17677669529663687f
#define EPS_    1e-5f

typedef __attribute__((ext_vector_type(16))) _Float16 v16h;
typedef __attribute__((ext_vector_type(8)))  float    v8f;

#define WMMA_F32_F16(a, b, c) \
  __builtin_amdgcn_wmma_f32_16x16x32_f16(false, (a), false, (b), (short)0, (c), false, false)

// ------------------------------------------------------- CDNA5 async copy
// Per-lane 16B global->LDS DMA, tracked by ASYNCcnt (gfx1250).
__device__ __forceinline__ void async_copy16(void* lds_dst, const void* gsrc) {
  asm volatile("global_load_async_to_lds_b128 %0, %1, off"
               :: "v"((uint32_t)(uintptr_t)lds_dst), "v"(gsrc)
               : "memory");
}
__device__ __forceinline__ void async_wait0() {
  asm volatile("s_wait_asynccnt 0x0" ::: "memory");
}

// ------------------------------------------------------- fragment loaders
// A (16x32 f16, MxK): lane m, one contiguous 16-half K run -> 2x ds_load_b128
__device__ __forceinline__ v16h frag_a(const _Float16* p, int ld, int row0, int col0, int lane) {
  int m = lane & 15, hi = lane >> 4;
  const _Float16* base = p + (size_t)(row0 + m) * ld + col0;
  v16h a;
#pragma unroll
  for (int v = 0; v < 8; ++v) {
    int k0 = hi * 16 + 2 * v;
    a[2 * v]     = base[k0];
    a[2 * v + 1] = base[k0 + 1];
  }
  return a;
}

// B (32x16 f16, KxN) from an N-major [n][k] store: lane n reads 16 contiguous
// halves -> 2x ds_load_b128
__device__ __forceinline__ v16h frag_b_tr(const _Float16* p, int ld, int k0, int n0, int lane) {
  int n = lane & 15, hi = lane >> 4;
  const _Float16* base = p + (size_t)(n0 + n) * ld + k0 + hi * 16;
  v16h b;
#pragma unroll
  for (int v = 0; v < 16; ++v) b[v] = base[v];
  return b;
}

// ---------------------------------------------------------------- kernels
// f32 [R][Cc] -> f16 transposed [Cc][R]
__global__ void k_f32_to_f16_t(const float* __restrict__ s, _Float16* __restrict__ d,
                               int R, int Cc) {
  int i = blockIdx.x * 256 + threadIdx.x;
  if (i < R * Cc) {
    int r = i / Cc, c = i % Cc;
    d[(size_t)c * R + r] = (_Float16)s[i];
  }
}

// LayerNorm1 + roll(-SS) + window partition -> f16 windows [NWIN][NTOK][CC]
__global__ __launch_bounds__(256) void k_ln1_window(
    const float* __restrict__ x, const float* __restrict__ g,
    const float* __restrict__ bt, _Float16* __restrict__ hw) {
  int lane = threadIdx.x & 31;
  int wave = __builtin_amdgcn_readfirstlane(threadIdx.x >> 5);
  int token = blockIdx.x * 8 + wave;                 // 131072 tokens
  const float4 v = *(const float4*)(x + (size_t)token * CC + lane * 4);
  float s = v.x + v.y + v.z + v.w;
#pragma unroll
  for (int m = 16; m >= 1; m >>= 1) s += __shfl_xor(s, m, 32);
  float mean = s * (1.0f / CC);
  float d0 = v.x - mean, d1 = v.y - mean, d2 = v.z - mean, d3 = v.w - mean;
  float vs = d0 * d0 + d1 * d1 + d2 * d2 + d3 * d3;
#pragma unroll
  for (int m = 16; m >= 1; m >>= 1) vs += __shfl_xor(vs, m, 32);
  float rstd = rsqrtf(vs * (1.0f / CC) + EPS_);
  int b = token >> 12, hw_ = token & 4095;
  int h = hw_ >> 6, w = hw_ & 63;
  int hs = (h + HH - SS_) & 63, ws = (w + HH - SS_) & 63;   // roll -SS
  int widx = b * 64 + (hs >> 3) * 8 + (ws >> 3);
  int n = (hs & 7) * 8 + (ws & 7);
  _Float16* pd = hw + ((size_t)widx * NTOK + n) * CC + lane * 4;
  int c0 = lane * 4;
  pd[0] = (_Float16)(d0 * rstd * g[c0 + 0] + bt[c0 + 0]);
  pd[1] = (_Float16)(d1 * rstd * g[c0 + 1] + bt[c0 + 1]);
  pd[2] = (_Float16)(d2 * rstd * g[c0 + 2] + bt[c0 + 2]);
  pd[3] = (_Float16)(d3 * rstd * g[c0 + 3] + bt[c0 + 3]);
}

__device__ __forceinline__ int region_id(int h) { return (h < HH - WS_) ? 0 : ((h < HH - SS_) ? 1 : 2); }

// Fused windowed attention.  GEMM orientations chosen so every epilogue store
// is contiguous in the D-tile row index (b128 stores); Q/K/V region selection
// is a scalar (SGPR) branch via readfirstlane-scalarized wave id.
__global__ __launch_bounds__(256) void k_attn(
    const _Float16* __restrict__ hw, const _Float16* __restrict__ wqkv_t,
    const float* __restrict__ bqkv, const _Float16* __restrict__ wproj_t,
    const float* __restrict__ bproj, const float* __restrict__ rpb,
    const float* __restrict__ x, float* __restrict__ x1) {
  extern __shared__ char smem[];
  _Float16* sWt  = (_Float16*)smem;                 //  96 KB  qkv W^T (384x128)
  _Float16* sX   = (_Float16*)(smem + 98304);       //  16 KB  x window / attn out
  _Float16* sQ   = (_Float16*)(smem + 114688);      //  16 KB  [tok][dim]
  _Float16* sK   = (_Float16*)(smem + 131072);      //  16 KB  [tok][dim]
  _Float16* sVt  = (_Float16*)(smem + 147456);      //  16 KB  [dim][tok]
  _Float16* sPWt = (_Float16*)(smem + 163840);      //  32 KB  proj W^T (128x128)
  float*    sS   = (float*)smem;                    //  64 KB  scores (reuse sWt)
  _Float16* sP   = (_Float16*)(smem + 65536);       //  32 KB  probs  (reuse sWt)
  float*    sRPB = (float*)(smem + 196608);         // 3600 B  rel-pos-bias table
  int*      sReg = (int*)(smem + 200208);           //  256 B  region id per token

  int widx = blockIdx.x, tid = threadIdx.x;
  int lane = tid & 31;
  int wave = __builtin_amdgcn_readfirstlane(tid >> 5);   // scalar wave id
  int nn = lane & 15, hi = lane >> 4;
  int b = widx >> 6, wi = widx & 63, wh = wi >> 3, ww = wi & 7;

  // ---- stage weights + window into LDS with async DMA (ASYNCcnt)
  {
    const uint4* gw = (const uint4*)wqkv_t;
    uint4* dw = (uint4*)sWt;
    for (int i = tid; i < (3 * CC * CC) / 8; i += 256) async_copy16(dw + i, gw + i);
    const uint4* gx = (const uint4*)(hw + (size_t)widx * NTOK * CC);
    uint4* dx = (uint4*)sX;
    for (int i = tid; i < (NTOK * CC) / 8; i += 256) async_copy16(dx + i, gx + i);
    const uint4* gp = (const uint4*)wproj_t;
    uint4* dp = (uint4*)sPWt;
    for (int i = tid; i < (CC * CC) / 8; i += 256) async_copy16(dp + i, gp + i);
  }
  for (int i = tid; i < 900; i += 256) sRPB[i] = rpb[i];
  if (tid < NTOK) {
    int i1 = tid >> 3, j1 = tid & 7;
    sReg[tid] = region_id(wh * 8 + i1) * 3 + region_id(ww * 8 + j1);
  }
  async_wait0();
  __syncthreads();

  // ---- QKV.  Q/K tiles flipped (rows = out-cols): contiguous sQ/sK stores.
  //      V tiles normal (rows = tokens): contiguous sVt stores.
#pragma unroll
  for (int i = 0; i < 8; ++i) {                      // 64 flipped Q/K tiles
    int t = wave * 8 + i;                            // scalar
    int cti = t >> 2, ttj = t & 3;                   // scalar col/tok tiles
    v8f acc = {};
#pragma unroll
    for (int kk = 0; kk < CC; kk += 32) {
      v16h a  = frag_a(sWt, CC, cti * 16, kk, lane);
      v16h bb = frag_b_tr(sX, CC, kk, ttj * 16, lane);
      acc = WMMA_F32_F16(a, bb, acc);
    }
    int tok = ttj * 16 + nn;
    int c0 = cti * 16 + 8 * hi;
    float4 ba = *(const float4*)(bqkv + c0);
    float4 bbq = *(const float4*)(bqkv + c0 + 4);
    if (cti < 8) {                                   // scalar branch: Q
      _Float16* d = sQ + tok * CC + c0;
#pragma unroll
      for (int r = 0; r < 4; ++r) d[r]     = (_Float16)((acc[r]     + (&ba.x)[r])  * SCALE_);
#pragma unroll
      for (int r = 0; r < 4; ++r) d[r + 4] = (_Float16)((acc[r + 4] + (&bbq.x)[r]) * SCALE_);
    } else {                                         // scalar branch: K
      _Float16* d = sK + tok * CC + (c0 - CC);
#pragma unroll
      for (int r = 0; r < 4; ++r) d[r]     = (_Float16)(acc[r]     + (&ba.x)[r]);
#pragma unroll
      for (int r = 0; r < 4; ++r) d[r + 4] = (_Float16)(acc[r + 4] + (&bbq.x)[r]);
    }
  }
#pragma unroll
  for (int i = 0; i < 4; ++i) {                      // 32 normal V tiles
    int t = wave * 4 + i;
    int vti = t >> 3, vtj = t & 7;                   // tok-tile, vcol-tile
    v8f acc = {};
#pragma unroll
    for (int kk = 0; kk < CC; kk += 32) {
      v16h a  = frag_a(sX, CC, vti * 16, kk, lane);
      v16h bb = frag_b_tr(sWt, CC, kk, 2 * CC + vtj * 16, lane);
      acc = WMMA_F32_F16(a, bb, acc);
    }
    int d = vtj * 16 + nn;                           // v-dim (lane)
    float bias = bqkv[2 * CC + d];
    _Float16* dst = sVt + d * NTOK + vti * 16 + 8 * hi;
#pragma unroll
    for (int r = 0; r < 8; ++r) dst[r] = (_Float16)(acc[r] + bias);
  }
  __syncthreads();

  // ---- scores: compute S^T = K @ Q^T per head -> contiguous f32 stores
#pragma unroll
  for (int i = 0; i < 8; ++i) {
    int t = wave * 8 + i;
    int head = t >> 4, rest = t & 15, kti = rest >> 2, qtj = rest & 3;
    v16h a  = frag_a(sK, CC, kti * 16, head * HD, lane);
    v16h bb = frag_b_tr(sQ, CC, head * HD, qtj * 16, lane);
    v8f acc = {};
    acc = WMMA_F32_F16(a, bb, acc);
    float* dst = sS + head * 4096 + (qtj * 16 + nn) * 64 + kti * 16 + 8 * hi;
    float4 f0 = {acc[0], acc[1], acc[2], acc[3]};
    float4 f1 = {acc[4], acc[5], acc[6], acc[7]};
    *(float4*)dst = f0;
    *(float4*)(dst + 4) = f1;
  }
  __syncthreads();

  // ---- softmax (+rpb +shift-mask fused): one row per thread
  {
    int head = tid >> 6, n = tid & 63;
    float* row = sS + head * 4096 + n * 64;
    int i1 = n >> 3, j1 = n & 7, reg1 = sReg[n];
    float mx = -1e30f;
#pragma unroll 8
    for (int m = 0; m < 64; ++m) {
      int ridx = (i1 - (m >> 3) + 7) * 15 + (j1 - (m & 7) + 7);
      float val = row[m] + sRPB[ridx * HEADS + head] +
                  ((sReg[m] == reg1) ? 0.0f : -100.0f);
      row[m] = val;
      mx = fmaxf(mx, val);
    }
    float sum = 0.0f;
#pragma unroll 8
    for (int m = 0; m < 64; ++m) { float e = __expf(row[m] - mx); row[m] = e; sum += e; }
    float inv = 1.0f / sum;
    _Float16* prow = sP + head * 4096 + n * 64;
#pragma unroll 8
    for (int m = 0; m < 64; ++m) prow[m] = (_Float16)(row[m] * inv);
  }
  __syncthreads();

  // ---- out^T = V^T @ P^T per head -> contiguous f16 stores into sX
#pragma unroll
  for (int i = 0; i < 4; ++i) {
    int t = wave * 4 + i;
    int head = t >> 3, rest = t & 7, dti = rest >> 2, qtj = rest & 3;
    v8f acc = {};
#pragma unroll
    for (int kk = 0; kk < NTOK; kk += 32) {
      v16h a  = frag_a(sVt, NTOK, head * HD + dti * 16, kk, lane);
      v16h bb = frag_b_tr(sP + head * 4096, 64, kk, qtj * 16, lane);
      acc = WMMA_F32_F16(a, bb, acc);
    }
    _Float16* dst = sX + (qtj * 16 + nn) * CC + head * HD + dti * 16 + 8 * hi;
#pragma unroll
    for (int r = 0; r < 8; ++r) dst[r] = (_Float16)acc[r];
  }
  __syncthreads();

  // ---- proj flipped (rows = out-cols) -> contiguous global float4 stores
#pragma unroll
  for (int i = 0; i < 4; ++i) {
    int t = wave * 4 + i;
    int cti = t >> 2, ttj = t & 3;
    v8f acc = {};
#pragma unroll
    for (int kk = 0; kk < CC; kk += 32) {
      v16h a  = frag_a(sPWt, CC, cti * 16, kk, lane);
      v16h bb = frag_b_tr(sX, CC, kk, ttj * 16, lane);
      acc = WMMA_F32_F16(a, bb, acc);
    }
    int tok = ttj * 16 + nn;
    int c0 = cti * 16 + 8 * hi;
    int inh = tok >> 3, inw = tok & 7;
    int h = (wh * 8 + inh + SS_) & 63;
    int w = (ww * 8 + inw + SS_) & 63;
    size_t idx = (((size_t)b << 12) + h * 64 + w) * CC + c0;
    float4 ba = *(const float4*)(bproj + c0);
    float4 bb4 = *(const float4*)(bproj + c0 + 4);
    float4 xa = *(const float4*)(x + idx);
    float4 xb = *(const float4*)(x + idx + 4);
    float4 o0 = {xa.x + acc[0] + ba.x, xa.y + acc[1] + ba.y,
                 xa.z + acc[2] + ba.z, xa.w + acc[3] + ba.w};
    float4 o1 = {xb.x + acc[4] + bb4.x, xb.y + acc[5] + bb4.y,
                 xb.z + acc[6] + bb4.z, xb.w + acc[7] + bb4.w};
    *(float4*)(x1 + idx) = o0;
    *(float4*)(x1 + idx + 4) = o1;
  }
}

// LayerNorm2 + FC1(flipped) + exact GELU + FC2(flipped) + residual.
// w1t = W1^T (512x128), w2t = W2^T (128x512)
__global__ __launch_bounds__(256) void k_mlp(
    const float* __restrict__ x1, const float* __restrict__ g,
    const float* __restrict__ bt, const _Float16* __restrict__ w1t,
    const float* __restrict__ b1, const _Float16* __restrict__ w2t,
    const float* __restrict__ b2, float* __restrict__ out) {
  extern __shared__ char smem[];
  _Float16* sH   = (_Float16*)smem;                 // 16 KB  normed input [tok][c]
  _Float16* sWp  = (_Float16*)(smem + 16384);       // 32 KB  weight panel [n][k]
  _Float16* sHid = (_Float16*)(smem + 49152);       // 64 KB  hidden [tok][512]

  int tid = threadIdx.x, lane = tid & 31;
  int wave = __builtin_amdgcn_readfirstlane(tid >> 5);   // scalar wave id
  int nn = lane & 15, hi = lane >> 4;
  size_t row0 = (size_t)blockIdx.x * 64;

  // LayerNorm2 (one token per wave iteration)
  for (int i = 0; i < 8; ++i) {
    int t = wave * 8 + i;
    const float4 v = *(const float4*)(x1 + (row0 + t) * CC + lane * 4);
    float s = v.x + v.y + v.z + v.w;
#pragma unroll
    for (int m = 16; m >= 1; m >>= 1) s += __shfl_xor(s, m, 32);
    float mean = s * (1.0f / CC);
    float d0 = v.x - mean, d1 = v.y - mean, d2 = v.z - mean, d3 = v.w - mean;
    float vs = d0 * d0 + d1 * d1 + d2 * d2 + d3 * d3;
#pragma unroll
    for (int m = 16; m >= 1; m >>= 1) vs += __shfl_xor(vs, m, 32);
    float rstd = rsqrtf(vs * (1.0f / CC) + EPS_);
    int c0 = lane * 4;
    sH[t * CC + c0 + 0] = (_Float16)(d0 * rstd * g[c0 + 0] + bt[c0 + 0]);
    sH[t * CC + c0 + 1] = (_Float16)(d1 * rstd * g[c0 + 1] + bt[c0 + 1]);
    sH[t * CC + c0 + 2] = (_Float16)(d2 * rstd * g[c0 + 2] + bt[c0 + 2]);
    sH[t * CC + c0 + 3] = (_Float16)(d3 * rstd * g[c0 + 3] + bt[c0 + 3]);
  }
  __syncthreads();

  // FC1 + GELU flipped: A = W1^T panel rows (hidden cols), B = sH gathered.
  // Contiguous stores into row-major sHid.
  for (int pj = 0; pj < 4; ++pj) {
    {
      const uint4* gw = (const uint4*)w1t + (size_t)pj * 2048;
      uint4* dw = (uint4*)sWp;
      for (int i = tid; i < 2048; i += 256) async_copy16(dw + i, gw + i);
    }
    async_wait0();
    __syncthreads();
#pragma unroll
    for (int i = 0; i < 4; ++i) {
      int t = wave * 4 + i;
      int hti = t >> 2, ttj = t & 3;                 // hid-col tile, tok tile
      v8f acc = {};
#pragma unroll
      for (int kk = 0; kk < CC; kk += 32) {
        v16h a  = frag_a(sWp, CC, hti * 16, kk, lane);
        v16h bb = frag_b_tr(sH, CC, kk, ttj * 16, lane);
        acc = WMMA_F32_F16(a, bb, acc);
      }
      int tok = ttj * 16 + nn;
      int hl0 = hti * 16 + 8 * hi;
      int colg = pj * CC + hl0;
      float4 ba = *(const float4*)(b1 + colg);
      float4 bb4 = *(const float4*)(b1 + colg + 4);
      _Float16* dst = sHid + tok * MLP_HID + colg;
#pragma unroll
      for (int r = 0; r < 8; ++r) {
        float v = acc[r] + ((r < 4) ? (&ba.x)[r] : (&bb4.x)[r - 4]);
        dst[r] = (_Float16)(0.5f * v * (1.0f + erff(v * 0.70710678f)));
      }
    }
    __syncthreads();
  }

  // FC2 flipped: A = W2^T panel rows (out cols), B = sHid gathered.
  // Contiguous global float4 stores with float4 residual loads.
  v8f acc[4];
#pragma unroll
  for (int i = 0; i < 4; ++i) acc[i] = (v8f){};
  for (int pk = 0; pk < 4; ++pk) {
    {
      const uint4* gw = (const uint4*)w2t;
      uint4* dw = (uint4*)sWp;
      for (int i = tid; i < 2048; i += 256) {
        int n = i >> 4, kq = i & 15;                 // 16 uint4 per 128-half row
        async_copy16(dw + n * 16 + kq, gw + n * 64 + pk * 16 + kq);
      }
    }
    async_wait0();
    __syncthreads();
#pragma unroll
    for (int i = 0; i < 4; ++i) {
      int t = wave * 4 + i;
      int cti = t >> 2, ttj = t & 3;
#pragma unroll
      for (int kk = 0; kk < CC; kk += 32) {
        v16h a  = frag_a(sWp, CC, cti * 16, kk, lane);
        v16h bb = frag_b_tr(sHid, MLP_HID, pk * CC + kk, ttj * 16, lane);
        acc[i] = WMMA_F32_F16(a, bb, acc[i]);
      }
    }
    __syncthreads();
  }
#pragma unroll
  for (int i = 0; i < 4; ++i) {
    int t = wave * 4 + i;
    int cti = t >> 2, ttj = t & 3;
    int tok = ttj * 16 + nn;
    int c0 = cti * 16 + 8 * hi;
    size_t idx = (row0 + tok) * CC + c0;
    float4 ba = *(const float4*)(b2 + c0);
    float4 bb4 = *(const float4*)(b2 + c0 + 4);
    float4 xa = *(const float4*)(x1 + idx);
    float4 xb = *(const float4*)(x1 + idx + 4);
    float4 o0 = {xa.x + acc[i][0] + ba.x, xa.y + acc[i][1] + ba.y,
                 xa.z + acc[i][2] + ba.z, xa.w + acc[i][3] + ba.w};
    float4 o1 = {xb.x + acc[i][4] + bb4.x, xb.y + acc[i][5] + bb4.y,
                 xb.z + acc[i][6] + bb4.z, xb.w + acc[i][7] + bb4.w};
    *(float4*)(out + idx) = o0;
    *(float4*)(out + idx + 4) = o1;
  }
}

// ---------------------------------------------------------------- launch
extern "C" void kernel_launch(void* const* d_in, const int* in_sizes, int n_in,
                              void* d_out, int out_size, void* d_ws, size_t ws_size,
                              hipStream_t stream) {
  const float* x     = (const float*)d_in[0];
  const float* n1g   = (const float*)d_in[1];
  const float* n1b   = (const float*)d_in[2];
  const float* qkvw  = (const float*)d_in[3];
  const float* qkvb  = (const float*)d_in[4];
  const float* projw = (const float*)d_in[5];
  const float* projb = (const float*)d_in[6];
  const float* rpb   = (const float*)d_in[7];
  const float* n2g   = (const float*)d_in[8];
  const float* n2b   = (const float*)d_in[9];
  const float* w1    = (const float*)d_in[10];
  const float* b1    = (const float*)d_in[11];
  const float* w2    = (const float*)d_in[12];
  const float* b2    = (const float*)d_in[13];
  float* out = (float*)d_out;

  char* ws = (char*)d_ws;
  _Float16* hw      = (_Float16*)ws;                      // 32 MB windows
  _Float16* wqkv_t  = (_Float16*)(ws + 33554432);         // 96 KB   (384x128)
  _Float16* wproj_t = (_Float16*)(ws + 33652736);         // 32 KB   (128x128)
  _Float16* w1_t    = (_Float16*)(ws + 33685504);         // 128 KB  (512x128)
  _Float16* w2_t    = (_Float16*)(ws + 33816576);         // 128 KB  (128x512)
  float*    x1      = (float*)(ws + 33947648);            // 64 MB residual-1

  k_f32_to_f16_t<<<(CC * 3 * CC + 255) / 256, 256, 0, stream>>>(qkvw, wqkv_t, CC, 3 * CC);
  k_f32_to_f16_t<<<(CC * CC + 255) / 256, 256, 0, stream>>>(projw, wproj_t, CC, CC);
  k_f32_to_f16_t<<<(CC * MLP_HID + 255) / 256, 256, 0, stream>>>(w1, w1_t, CC, MLP_HID);
  k_f32_to_f16_t<<<(MLP_HID * CC + 255) / 256, 256, 0, stream>>>(w2, w2_t, MLP_HID, CC);

  k_ln1_window<<<16384, 256, 0, stream>>>(x, n1g, n1b, hw);
  k_attn<<<NWIN, 256, 200704, stream>>>(hw, wqkv_t, qkvb, wproj_t, projb, rpb, x, x1);
  k_mlp<<<2048, 256, 114688, stream>>>(x1, n2g, n2b, w1_t, b1, w2_t, b2, out);
}
